// GatedGRU_13013750907453
// MI455X (gfx1250) — compile-verified, matrix operand order checked by reading, not run on hardware
//
#include <hip/hip_runtime.h>
#include <hip/hip_bf16.h>

#define V_SZ 50000
#define E_SZ 1024
#define H_SZ 1024
#define O_SZ 50000
#define B_SZ 64
#define S_SZ 512

#define SCAN_BLOCKS 16           // persistent scan grid: 16 blocks x 8 waves = 128 waves

typedef __bf16 bf16_t;
typedef bf16_t v16bf __attribute__((ext_vector_type(16)));
typedef float  v8f   __attribute__((ext_vector_type(8)));

// ---------- helpers ----------

__device__ __forceinline__ bf16_t f2bf(float f) {
  union { float f; unsigned u; } v; v.f = f;
  unsigned u = v.u;
  u += 0x7FFFu + ((u >> 16) & 1u);          // round-to-nearest-even
  union { unsigned short s; bf16_t b; } r;
  r.s = (unsigned short)(u >> 16);
  return r.b;
}

// WMMA 16-bit operand fragment for this lane: 8 bf16 at p[0..7] (K = k0..k0+7)
// and 8 bf16 at p[16..23] (K = k0+16..k0+23).  Two 16-byte loads.
__device__ __forceinline__ v16bf load_frag(const bf16_t* p) {
  union { uint4 u[2]; v16bf v; } w;
  w.u[0] = *(const uint4*)(p);
  w.u[1] = *(const uint4*)(p + 16);
  return w.v;
}

__device__ __forceinline__ float sigmoidf_(float x) {
  return 1.0f / (1.0f + __expf(-x));
}

// Cooperative A-tile staging: 256 threads copy 16 rows x 1024 bf16 (32 KB) to LDS.
// Thread (row = tid>>4, seg = tid&15) moves one 128-byte segment of its row.
__device__ __forceinline__ void copy_row_seg(bf16_t* dst, const bf16_t* src) {
  const int seg = threadIdx.x & 15;
  const uint4* s = (const uint4*)(src + seg * 64);
  uint4* d = (uint4*)(dst + seg * 64);
#pragma unroll
  for (int i = 0; i < 8; ++i) d[i] = s[i];
}

// Device-wide barrier: one arrival per block, monotonic counter zeroed per launch.
__device__ __forceinline__ void grid_barrier(unsigned* ctr, unsigned target) {
  __syncthreads();
  if (threadIdx.x == 0) {
    __hip_atomic_fetch_add(ctr, 1u, __ATOMIC_ACQ_REL, __HIP_MEMORY_SCOPE_AGENT);
    unsigned v;
    do {
      __builtin_amdgcn_s_sleep(2);
      v = __hip_atomic_load(ctr, __ATOMIC_ACQUIRE, __HIP_MEMORY_SCOPE_AGENT);
    } while (v < target);
  }
  __syncthreads();
}

__global__ void init_ctr_kernel(unsigned* ctr) {
  if (threadIdx.x == 0 && blockIdx.x == 0) *ctr = 0u;
}

// ---------- elementwise f32 -> bf16 convert (8 elems / thread) ----------
__global__ void cvt_kernel(const float* __restrict__ in,
                           bf16_t* __restrict__ out, size_t n) {
  size_t i = ((size_t)blockIdx.x * blockDim.x + threadIdx.x) * 8;
  if (i >= n) return;
  union { float4 v[2]; float f[8]; } u;
  u.v[0] = *(const float4*)(in + i);
  u.v[1] = *(const float4*)(in + i + 4);
  union { uint4 v; unsigned short s[8]; } o;
#pragma unroll
  for (int j = 0; j < 8; ++j) {
    union { unsigned short s; bf16_t b; } c; c.b = f2bf(u.f[j]);
    o.s[j] = c.s;
  }
  *(uint4*)(out + i) = o.v;
}

// ---------- LDS-tiled weight transpose + f32 -> bf16 ----------
// in: (K, N) row-major f32  ->  out: (N, K) row-major bf16; coalesced both sides.
__global__ void cvt_transpose_kernel(const float* __restrict__ in,
                                     bf16_t* __restrict__ out,
                                     int K, int N) {
  __shared__ bf16_t tile[32][33];
  const int nb = blockIdx.x * 32;
  const int kb = blockIdx.y * 32;
  const int tx = threadIdx.x & 31;
  const int ty = threadIdx.x >> 5;          // 0..7
#pragma unroll
  for (int i = ty; i < 32; i += 8) {
    const int k = kb + i, n = nb + tx;
    if (k < K && n < N) tile[i][tx] = f2bf(in[(size_t)k * N + n]);
  }
  __syncthreads();
#pragma unroll
  for (int i = ty; i < 32; i += 8) {
    const int n = nb + i, k = kb + tx;
    if (n < N && k < K) out[(size_t)n * K + k] = tile[tx][i];
  }
}

// ---------- input projections: X_g = gather(emb_bf, tok) @ W_ig + b_g ----------
// Block stages its shared 16-row A-tile (gathered embeddings) in LDS once;
// each of 8 waves computes a 16(M) x 64(N) strip: 4 accumulators.
__global__ void proj_kernel(const int* __restrict__ tok,
                            const bf16_t* __restrict__ emb_bf,  // (V, E) bf16
                            const bf16_t* __restrict__ WiT,     // 3 x (H, E) bf16 T
                            const float* __restrict__ b_u,
                            const float* __restrict__ b_r,
                            const float* __restrict__ b_h,
                            float* __restrict__ X) {            // 3 x (B*S, H) f32
  __shared__ bf16_t Atile[16 * E_SZ];

  const int lane = threadIdx.x & 31;
  const int wave = threadIdx.x >> 5;
  const int g    = blockIdx.z;
  const int m0   = blockIdx.x * 16;
  const int nb   = (blockIdx.y * 8 + wave) * 64;
  const int mr   = lane & 15;
  const int hi   = lane >> 4;

  const bf16_t* WT = WiT + (size_t)g * H_SZ * E_SZ;
  float* Xg = X + (size_t)g * (size_t)B_SZ * S_SZ * H_SZ;
  const float* bias = (g == 0) ? b_u : (g == 1 ? b_r : b_h);

  {
    const int arow_id = threadIdx.x >> 4;     // 0..15
    const int t_id = tok[m0 + arow_id];
    copy_row_seg(Atile + arow_id * E_SZ, emb_bf + (size_t)t_id * E_SZ);
  }
  __syncthreads();

  const bf16_t* alds  = Atile + mr * E_SZ;
  const bf16_t* brow0 = WT + (size_t)(nb +  0 + mr) * E_SZ;
  const bf16_t* brow1 = WT + (size_t)(nb + 16 + mr) * E_SZ;
  const bf16_t* brow2 = WT + (size_t)(nb + 32 + mr) * E_SZ;
  const bf16_t* brow3 = WT + (size_t)(nb + 48 + mr) * E_SZ;

  v8f acc0 = {}, acc1 = {}, acc2 = {}, acc3 = {};
#pragma unroll 2
  for (int k = 0; k < E_SZ; k += 32) {
    const int ka = k + hi * 8;
    __builtin_prefetch(brow0 + ka + 256, 0, 1);
    v16bf a = load_frag(alds + ka);
    acc0 = __builtin_amdgcn_wmma_f32_16x16x32_bf16(false, a, false, load_frag(brow0 + ka), (short)0, acc0, false, false);
    acc1 = __builtin_amdgcn_wmma_f32_16x16x32_bf16(false, a, false, load_frag(brow1 + ka), (short)0, acc1, false, false);
    acc2 = __builtin_amdgcn_wmma_f32_16x16x32_bf16(false, a, false, load_frag(brow2 + ka), (short)0, acc2, false, false);
    acc3 = __builtin_amdgcn_wmma_f32_16x16x32_bf16(false, a, false, load_frag(brow3 + ka), (short)0, acc3, false, false);
  }

#pragma unroll
  for (int s = 0; s < 4; ++s) {
    const v8f acc = (s == 0) ? acc0 : (s == 1) ? acc1 : (s == 2) ? acc2 : acc3;
    const int n_l = nb + s * 16 + mr;
    const float bv = bias[n_l];
#pragma unroll
    for (int j = 0; j < 8; ++j) {
      const int m = m0 + hi * 8 + j;
      Xg[(size_t)m * H_SZ + n_l] = acc[j] + bv;
    }
  }
}

// ---------- persistent GRU scan over all 512 timesteps ----------
// 16 blocks x 8 waves = 128 waves.  (g, mtile) is block-uniform, so the block's
// 16-row A-tile is staged in LDS once per phase.  Per step:
//   phase 1 (128 waves): z and r gates, each wave a 16x64 strip; r*h -> rh_bf
//   phase 2 ( 64 waves): candidate tanh GEMM + blend; h -> hs (f32) + h_bf (bf16)
__global__ void gru_scan_kernel(const float* __restrict__ X,
                                const bf16_t* __restrict__ WhT,   // 3 x (H, H) bf16 T
                                const bf16_t* __restrict__ h0_bf,
                                const float* __restrict__ init_state,
                                float* __restrict__ zs, float* __restrict__ rs,
                                float* __restrict__ hs,
                                float* __restrict__ hT, bf16_t* __restrict__ hT_bf,
                                bf16_t* __restrict__ h_bf, bf16_t* __restrict__ rh_bf,
                                unsigned* __restrict__ ctr) {
  __shared__ bf16_t Atile[16 * H_SZ];

  const int lane = threadIdx.x & 31;
  const int wave = threadIdx.x >> 5;
  const int gw   = blockIdx.x * 8 + wave;        // 0..127
  const int mr   = lane & 15;
  const int hi   = lane >> 4;
  unsigned gen = 0;

  for (int t = 0; t < S_SZ; ++t) {
    const bf16_t* hp_bf = (t == 0) ? h0_bf : h_bf;
    const float*  hp_f  = (t == 0) ? init_state : (hs + (size_t)(t - 1) * B_SZ * H_SZ);
    const int hv = (t == 0) ? 1 : 0;

    // ---------------- phase 1: z and r gates ----------------
    {
      const int g      = gw >> 6;                // 0 -> z, 1 -> r  (block-uniform)
      const int mtile  = (gw >> 4) & 3;          // block-uniform
      const int nstrip = gw & 15;
      const int m0 = mtile * 16;
      const int nb = nstrip * 64;

      // stage h_prev rows m0..m0+15 into LDS (broadcast row if t==0)
      {
        const int arow_id = threadIdx.x >> 4;
        const bf16_t* src = hv ? hp_bf : (hp_bf + (size_t)(m0 + arow_id) * H_SZ);
        copy_row_seg(Atile + arow_id * H_SZ, src);
      }
      __syncthreads();

      const float*  Xg = X + (size_t)g * (size_t)B_SZ * S_SZ * H_SZ;
      const bf16_t* WT = WhT + (size_t)g * H_SZ * H_SZ;

      const bf16_t* alds  = Atile + mr * H_SZ;
      const bf16_t* brow0 = WT + (size_t)(nb +  0 + mr) * H_SZ;
      const bf16_t* brow1 = WT + (size_t)(nb + 16 + mr) * H_SZ;
      const bf16_t* brow2 = WT + (size_t)(nb + 32 + mr) * H_SZ;
      const bf16_t* brow3 = WT + (size_t)(nb + 48 + mr) * H_SZ;

      v8f acc0, acc1, acc2, acc3;
#pragma unroll
      for (int j = 0; j < 8; ++j) {
        const int b = m0 + hi * 8 + j;
        const size_t row = ((size_t)b * S_SZ + t) * H_SZ;
        acc0[j] = Xg[row + nb +  0 + mr];
        acc1[j] = Xg[row + nb + 16 + mr];
        acc2[j] = Xg[row + nb + 32 + mr];
        acc3[j] = Xg[row + nb + 48 + mr];
      }

#pragma unroll 2
      for (int k = 0; k < H_SZ; k += 32) {
        const int ka = k + hi * 8;
        v16bf a = load_frag(alds + ka);
        acc0 = __builtin_amdgcn_wmma_f32_16x16x32_bf16(false, a, false, load_frag(brow0 + ka), (short)0, acc0, false, false);
        acc1 = __builtin_amdgcn_wmma_f32_16x16x32_bf16(false, a, false, load_frag(brow1 + ka), (short)0, acc1, false, false);
        acc2 = __builtin_amdgcn_wmma_f32_16x16x32_bf16(false, a, false, load_frag(brow2 + ka), (short)0, acc2, false, false);
        acc3 = __builtin_amdgcn_wmma_f32_16x16x32_bf16(false, a, false, load_frag(brow3 + ka), (short)0, acc3, false, false);
      }

      float* trace = (g == 0) ? zs : rs;
#pragma unroll
      for (int s = 0; s < 4; ++s) {
        const v8f acc = (s == 0) ? acc0 : (s == 1) ? acc1 : (s == 2) ? acc2 : acc3;
        const int n_l = nb + s * 16 + mr;
#pragma unroll
        for (int j = 0; j < 8; ++j) {
          const int b = m0 + hi * 8 + j;
          const float v = sigmoidf_(acc[j]);
          trace[((size_t)t * B_SZ + b) * H_SZ + n_l] = v;
          if (g == 1) {
            const float hpv = hv ? hp_f[n_l] : hp_f[(size_t)b * H_SZ + n_l];
            rh_bf[(size_t)b * H_SZ + n_l] = f2bf(v * hpv);
          }
        }
      }
    }

    gen++; grid_barrier(ctr, gen * SCAN_BLOCKS);

    // ---------------- phase 2: candidate + blend ----------------
    if (gw < 64) {                               // block-uniform predicate
      const int mtile  = gw >> 4;
      const int nstrip = gw & 15;
      const int m0 = mtile * 16;
      const int nb = nstrip * 64;

      {
        const int arow_id = threadIdx.x >> 4;
        copy_row_seg(Atile + arow_id * H_SZ, rh_bf + (size_t)(m0 + arow_id) * H_SZ);
      }
      __syncthreads();

      const float*  Xg = X + (size_t)2 * (size_t)B_SZ * S_SZ * H_SZ;
      const bf16_t* WT = WhT + (size_t)2 * H_SZ * H_SZ;

      const bf16_t* alds  = Atile + mr * H_SZ;
      const bf16_t* brow0 = WT + (size_t)(nb +  0 + mr) * H_SZ;
      const bf16_t* brow1 = WT + (size_t)(nb + 16 + mr) * H_SZ;
      const bf16_t* brow2 = WT + (size_t)(nb + 32 + mr) * H_SZ;
      const bf16_t* brow3 = WT + (size_t)(nb + 48 + mr) * H_SZ;

      v8f acc0, acc1, acc2, acc3;
#pragma unroll
      for (int j = 0; j < 8; ++j) {
        const int b = m0 + hi * 8 + j;
        const size_t row = ((size_t)b * S_SZ + t) * H_SZ;
        acc0[j] = Xg[row + nb +  0 + mr];
        acc1[j] = Xg[row + nb + 16 + mr];
        acc2[j] = Xg[row + nb + 32 + mr];
        acc3[j] = Xg[row + nb + 48 + mr];
      }

#pragma unroll 2
      for (int k = 0; k < H_SZ; k += 32) {
        const int ka = k + hi * 8;
        v16bf a = load_frag(alds + ka);
        acc0 = __builtin_amdgcn_wmma_f32_16x16x32_bf16(false, a, false, load_frag(brow0 + ka), (short)0, acc0, false, false);
        acc1 = __builtin_amdgcn_wmma_f32_16x16x32_bf16(false, a, false, load_frag(brow1 + ka), (short)0, acc1, false, false);
        acc2 = __builtin_amdgcn_wmma_f32_16x16x32_bf16(false, a, false, load_frag(brow2 + ka), (short)0, acc2, false, false);
        acc3 = __builtin_amdgcn_wmma_f32_16x16x32_bf16(false, a, false, load_frag(brow3 + ka), (short)0, acc3, false, false);
      }

#pragma unroll
      for (int s = 0; s < 4; ++s) {
        const v8f acc = (s == 0) ? acc0 : (s == 1) ? acc1 : (s == 2) ? acc2 : acc3;
        const int n_l = nb + s * 16 + mr;
#pragma unroll
        for (int j = 0; j < 8; ++j) {
          const int b = m0 + hi * 8 + j;
          const float th = tanhf(acc[j]);
          const float z  = zs[((size_t)t * B_SZ + b) * H_SZ + n_l];
          const float hpv = hv ? hp_f[n_l] : hp_f[(size_t)b * H_SZ + n_l];
          const float hn = (1.0f - z) * hpv + z * th;
          hs[((size_t)t * B_SZ + b) * H_SZ + n_l] = hn;
          h_bf[(size_t)b * H_SZ + n_l] = f2bf(hn);
          if (t == S_SZ - 1) {
            hT[(size_t)b * H_SZ + n_l] = hn;
            hT_bf[(size_t)b * H_SZ + n_l] = f2bf(hn);
          }
        }
      }
    }

    gen++; grid_barrier(ctr, gen * SCAN_BLOCKS);
  }
}

// ---------- decoder: out = h_T @ dec_W + dec_b ----------
__global__ void decoder_kernel(const bf16_t* __restrict__ hT_bf,
                               const bf16_t* __restrict__ decWT,  // (O, H) bf16 T
                               const float* __restrict__ dec_b,
                               float* __restrict__ out) {
  __shared__ bf16_t Atile[16 * H_SZ];

  const int lane = threadIdx.x & 31;
  const int wave = threadIdx.x >> 5;
  const int m0   = blockIdx.x * 16;
  const int ngrp = blockIdx.y * 8 + wave;
  const int NT   = O_SZ / 16;                    // 3125 tiles
  const int mr   = lane & 15;
  const int hi   = lane >> 4;

  {
    const int arow_id = threadIdx.x >> 4;
    copy_row_seg(Atile + arow_id * H_SZ, hT_bf + (size_t)(m0 + arow_id) * H_SZ);
  }
  __syncthreads();

  const int nt0 = ngrp * 4;
  const int nt_c0 = (nt0 + 0 < NT) ? nt0 + 0 : NT - 1;
  const int nt_c1 = (nt0 + 1 < NT) ? nt0 + 1 : NT - 1;
  const int nt_c2 = (nt0 + 2 < NT) ? nt0 + 2 : NT - 1;
  const int nt_c3 = (nt0 + 3 < NT) ? nt0 + 3 : NT - 1;

  const bf16_t* alds  = Atile + mr * H_SZ;
  const bf16_t* brow0 = decWT + (size_t)(nt_c0 * 16 + mr) * H_SZ;
  const bf16_t* brow1 = decWT + (size_t)(nt_c1 * 16 + mr) * H_SZ;
  const bf16_t* brow2 = decWT + (size_t)(nt_c2 * 16 + mr) * H_SZ;
  const bf16_t* brow3 = decWT + (size_t)(nt_c3 * 16 + mr) * H_SZ;

  v8f acc0 = {}, acc1 = {}, acc2 = {}, acc3 = {};
#pragma unroll 2
  for (int k = 0; k < H_SZ; k += 32) {
    const int ka = k + hi * 8;
    __builtin_prefetch(brow0 + ka + 256, 0, 1);
    __builtin_prefetch(brow2 + ka + 256, 0, 1);
    v16bf a = load_frag(alds + ka);
    acc0 = __builtin_amdgcn_wmma_f32_16x16x32_bf16(false, a, false, load_frag(brow0 + ka), (short)0, acc0, false, false);
    acc1 = __builtin_amdgcn_wmma_f32_16x16x32_bf16(false, a, false, load_frag(brow1 + ka), (short)0, acc1, false, false);
    acc2 = __builtin_amdgcn_wmma_f32_16x16x32_bf16(false, a, false, load_frag(brow2 + ka), (short)0, acc2, false, false);
    acc3 = __builtin_amdgcn_wmma_f32_16x16x32_bf16(false, a, false, load_frag(brow3 + ka), (short)0, acc3, false, false);
  }

#pragma unroll
  for (int s = 0; s < 4; ++s) {
    const int nt = nt0 + s;
    if (nt >= NT) break;
    const v8f acc = (s == 0) ? acc0 : (s == 1) ? acc1 : (s == 2) ? acc2 : acc3;
    const int n_l = nt * 16 + mr;
    const float bv = dec_b[n_l];
#pragma unroll
    for (int j = 0; j < 8; ++j) {
      const int m = m0 + hi * 8 + j;
      out[(size_t)m * O_SZ + n_l] = acc[j] + bv;
    }
  }
}

// ---------- host launcher ----------
extern "C" void kernel_launch(void* const* d_in, const int* in_sizes, int n_in,
                              void* d_out, int out_size, void* d_ws, size_t ws_size,
                              hipStream_t stream) {
  (void)in_sizes; (void)n_in; (void)out_size; (void)ws_size;

  const int*   x          = (const int*)  d_in[0];
  const float* init_state = (const float*)d_in[1];
  const float* embedding  = (const float*)d_in[2];
  const float* W_iu  = (const float*)d_in[3];
  const float* W_hu  = (const float*)d_in[4];
  const float* b_u   = (const float*)d_in[5];
  const float* W_ir  = (const float*)d_in[6];
  const float* W_hr  = (const float*)d_in[7];
  const float* b_r   = (const float*)d_in[8];
  const float* W_ih  = (const float*)d_in[9];
  const float* W_hh  = (const float*)d_in[10];
  const float* b_h   = (const float*)d_in[11];
  const float* dec_W = (const float*)d_in[12];
  const float* dec_b = (const float*)d_in[13];

  // workspace carve (all 16B aligned)
  char* w = (char*)d_ws;
  bf16_t* WiT    = (bf16_t*)w;  w += (size_t)3 * H_SZ * E_SZ * sizeof(bf16_t);
  bf16_t* WhT    = (bf16_t*)w;  w += (size_t)3 * H_SZ * H_SZ * sizeof(bf16_t);
  bf16_t* decWT  = (bf16_t*)w;  w += (size_t)O_SZ * H_SZ * sizeof(bf16_t);
  bf16_t* emb_bf = (bf16_t*)w;  w += (size_t)V_SZ * E_SZ * sizeof(bf16_t);
  float*  X      = (float*)w;   w += (size_t)3 * B_SZ * S_SZ * H_SZ * sizeof(float);
  bf16_t* h_bf   = (bf16_t*)w;  w += (size_t)B_SZ * H_SZ * sizeof(bf16_t);
  bf16_t* rh_bf  = (bf16_t*)w;  w += (size_t)B_SZ * H_SZ * sizeof(bf16_t);
  bf16_t* hT_bf  = (bf16_t*)w;  w += (size_t)B_SZ * H_SZ * sizeof(bf16_t);
  bf16_t* h0_bf  = (bf16_t*)w;  w += (size_t)H_SZ * sizeof(bf16_t);
  unsigned* ctr  = (unsigned*)w; w += 256;

  // 0) zero the grid-barrier counter (keeps the launch deterministic)
  init_ctr_kernel<<<dim3(1), dim3(64), 0, stream>>>(ctr);

  // 1) conversions
  {
    size_t n = (size_t)V_SZ * E_SZ;
    int blocks = (int)((n / 8 + 255) / 256);
    cvt_kernel<<<dim3(blocks), dim3(256), 0, stream>>>(embedding, emb_bf, n);
    cvt_kernel<<<dim3(1), dim3(256), 0, stream>>>(init_state, h0_bf, (size_t)H_SZ);
  }
  auto tlaunch = [&](const float* in, bf16_t* out, int K, int N) {
    cvt_transpose_kernel<<<dim3((N + 31) / 32, (K + 31) / 32), dim3(256), 0, stream>>>(
        in, out, K, N);
  };
  tlaunch(W_iu, WiT + (size_t)0 * H_SZ * E_SZ, E_SZ, H_SZ);
  tlaunch(W_ir, WiT + (size_t)1 * H_SZ * E_SZ, E_SZ, H_SZ);
  tlaunch(W_ih, WiT + (size_t)2 * H_SZ * E_SZ, E_SZ, H_SZ);
  tlaunch(W_hu, WhT + (size_t)0 * H_SZ * H_SZ, H_SZ, H_SZ);
  tlaunch(W_hr, WhT + (size_t)1 * H_SZ * H_SZ, H_SZ, H_SZ);
  tlaunch(W_hh, WhT + (size_t)2 * H_SZ * H_SZ, H_SZ, H_SZ);
  tlaunch(dec_W, decWT, H_SZ, O_SZ);

  // 2) input projections (big WMMA GEMM, A-tile staged in LDS per block)
  proj_kernel<<<dim3((B_SZ * S_SZ) / 16, 2, 3), 256, 0, stream>>>(
      x, emb_bf, WiT, b_u, b_r, b_h, X);

  // d_out carve: output, h_T, zs, rs, hs (reference return order)
  float* out_logits = (float*)d_out;
  float* hT = out_logits + (size_t)B_SZ * O_SZ;
  float* zs = hT + (size_t)B_SZ * H_SZ;
  float* rs = zs + (size_t)S_SZ * B_SZ * H_SZ;
  float* hs = rs + (size_t)S_SZ * B_SZ * H_SZ;

  // 3) whole recurrence in ONE persistent kernel (1024 launches -> 1)
  gru_scan_kernel<<<dim3(SCAN_BLOCKS), 256, 0, stream>>>(
      X, WhT, h0_bf, init_state, zs, rs, hs, hT, hT_bf, h_bf, rh_bf, ctr);

  // 4) decoder (N = 50000 = 3125 tiles; 4 tiles per wave)
  decoder_kernel<<<dim3(B_SZ / 16, ((O_SZ / 16 + 3) / 4 + 7) / 8), 256, 0, stream>>>(
      hT_bf, decWT, dec_b, out_logits);
}
